// AttentiveMMDPrompt_29729763623085
// MI455X (gfx1250) — matrix operand order repaired and checked by reading, not compiled, not run on hardware
//
#include <hip/hip_runtime.h>
#include <hip/hip_bf16.h>

// ---------------------------------------------------------------------------
// AttentiveMMDPrompt on MI455X (gfx1250, wave32).
// Dominant cost: l = local_f @ Wk^T  (117600 x 640 x 640 = 96.3 GFLOP)
//   -> v_wmma_f32_16x16x32_bf16, fp32 accumulation (compute-bound; scalar f32
//      VALU would be ~20x slower, f32 WMMA carries 8x less K per issue).
// local_f = 301 MB fp32 -> ~13us/pass at 23.3 TB/s. l is staged once in d_ws
// (store+reload ~26us < recomputing the GEMM ~39us at ~2.5 PF/s bf16).
// Per-batch token sums for the centering mean are fused into the GEMM
// epilogue. Streaming traffic uses non-temporal hints so Wk stays in L2.
// ---------------------------------------------------------------------------

#define B_    8
#define NG_   5
#define NL_   75
#define NF_   196
#define C_    640
#define M_TOTAL   (B_ * NL_ * NF_)   // 117600 rows of the big GEMM
#define MB_ROWS   (NL_ * NF_)        // 14700 rows per batch
#define NTOK      (NG_ + NL_ * NF_)  // 14705 tokens per batch (mean denom)
#define ALPHA_INV 10.0f
#define EPS_      1e-12f

typedef __attribute__((ext_vector_type(16))) __bf16 v16bf;
typedef __attribute__((ext_vector_type(8)))  __bf16 v8bf;
typedef __attribute__((ext_vector_type(8)))  float  v8f;
typedef __attribute__((ext_vector_type(4)))  float  v4f;

// Native RTNE convert: clang lowers __bf16 casts to the hardware cvt ops
// (far leaner than integer round emulation).
__device__ __forceinline__ unsigned int pack_bf16(float x, float y) {
    unsigned short a = __builtin_bit_cast(unsigned short, (__bf16)x);
    unsigned short b = __builtin_bit_cast(unsigned short, (__bf16)y);
    return ((unsigned int)b << 16) | a;
}

// ---------------------------------------------------------------------------
__global__ void zero_kernel(float* __restrict__ p, int n) {
    int i = blockIdx.x * blockDim.x + threadIdx.x;
    if (i < n) p[i] = 0.0f;
}

// ---------------------------------------------------------------------------
// g = global_f @ Wq^T  (40 rows; trivial). Also accumulate per-batch sums.
__global__ void __launch_bounds__(128)
proj_global_kernel(const float* __restrict__ gf, const float* __restrict__ Wq,
                   float* __restrict__ gproj, float* __restrict__ sum_g) {
    const int blk = blockIdx.x;            // b*NG + gi
    const int b = blk / NG_;
    __shared__ float xs[C_];
    const int tid = threadIdx.x;
    for (int c = tid; c < C_; c += 128) xs[c] = gf[(size_t)blk * C_ + c];
    __syncthreads();
#pragma unroll
    for (int j = 0; j < 5; ++j) {
        const int d = tid + j * 128;
        const float* wr = Wq + (size_t)d * C_;
        float acc = 0.0f;
        for (int c = 0; c < C_; ++c) acc = fmaf(xs[c], wr[c], acc);
        gproj[(size_t)blk * C_ + d] = acc;
        atomicAdd(&sum_g[b * C_ + d], acc);
    }
}

// ---------------------------------------------------------------------------
// Heavy GEMM: l[m,d] = sum_c local_f[m,c] * Wk[d,c]
// Block tile 128(M) x 64(N), 8 waves; wave tile 16 x 64 (4 WMMA accums).
// K staged 64-wide in LDS as bf16 (8 v_wmma per wave per barrier window).
// Row stride 72 bf16 = 144 B = 36 banks -> conflict-free fragment reads.
__global__ void __launch_bounds__(256)
proj_local_wmma_kernel(const float* __restrict__ A, const float* __restrict__ Wk,
                       float* __restrict__ Lout, float* __restrict__ sum_l) {
    __shared__ __align__(16) unsigned short ldsA[128 * 72];
    __shared__ __align__(16) unsigned short ldsB[64 * 72];
    __shared__ float colSum[128];          // [2 batches][64 cols]

    const int tid   = threadIdx.x;
    const int Mbase = blockIdx.x * 128;
    const int Nbase = blockIdx.y * 64;
    const int wave  = tid >> 5;
    const int lane  = tid & 31;
    const int waveM = wave * 16;

    v8f acc[4];
#pragma unroll
    for (int s = 0; s < 4; ++s) acc[s] = (v8f)0.0f;

    const int am = lane & 15;

    for (int kk = 0; kk < C_; kk += 64) {
        // ---- stage A tile: 128 rows x 64 K (fp32 -> bf16, streaming) ----
#pragma unroll
        for (int i = 0; i < 8; ++i) {
            const int e4  = tid + i * 256;          // 2048 float4 loads
            const int row = e4 >> 4;                // 16 float4 per row
            const int col = (e4 & 15) << 2;
            // clamp instead of branch: OOB rows (last M block only) load a
            // valid duplicate row; they are excluded at store/sum time.
            const int grow = min(Mbase + row, M_TOTAL - 1);
            v4f v = __builtin_nontemporal_load(
                (const v4f*)(A + (size_t)grow * C_ + kk + col));
            *(uint2*)&ldsA[row * 72 + col] =
                make_uint2(pack_bf16(v[0], v[1]), pack_bf16(v[2], v[3]));
        }
        // ---- stage B tile: Wk rows [Nbase,Nbase+64) x 64 K (cached) ----
#pragma unroll
        for (int i = 0; i < 4; ++i) {
            const int e4  = tid + i * 256;          // 1024 float4 loads
            const int row = e4 >> 4;
            const int col = (e4 & 15) << 2;
            v4f v = *(const v4f*)(Wk + (size_t)(Nbase + row) * C_ + kk + col);
            *(uint2*)&ldsB[row * 72 + col] =
                make_uint2(pack_bf16(v[0], v[1]), pack_bf16(v[2], v[3]));
        }
        __syncthreads();

#pragma unroll
        for (int c0 = 0; c0 < 64; c0 += 32) {
            // A fragment (ISA 16-bit A 16x32 layout)
            const int kb = c0 + ((lane >> 4) << 3);     // +0 or +8
            v8bf alo = *(const v8bf*)&ldsA[(waveM + am) * 72 + kb];
            v8bf ahi = *(const v8bf*)&ldsA[(waveM + am) * 72 + kb + 16];
            v16bf afrag = __builtin_shufflevector(alo, ahi,
                0,1,2,3,4,5,6,7,8,9,10,11,12,13,14,15);

            // B fragments + WMMA (B[k][n] = Wk[n][k])
            const int bkb = c0 + ((lane >> 4) << 3);    // +0 or +8
#pragma unroll
            for (int s = 0; s < 4; ++s) {
                const unsigned short* bp = &ldsB[(s * 16 + am) * 72 + bkb];
                v8bf b0 = *(const v8bf*)bp;
                v8bf b1 = *(const v8bf*)(bp + 16);
                v16bf bfrag = __builtin_shufflevector(b0, b1,
                    0,1,2,3,4,5,6,7,8,9,10,11,12,13,14,15);
                acc[s] = __builtin_amdgcn_wmma_f32_16x16x32_bf16(
                    false, afrag, false, bfrag, (short)0, acc[s], false, false);
            }
        }
        __syncthreads();
    }

    // ---- epilogue: store D (streaming), fold per-batch column sums ----
    if (tid < 128) colSum[tid] = 0.0f;
    __syncthreads();

    const int b0       = Mbase / MB_ROWS;                 // <=2 batches/block
    const int mrowBase = Mbase + waveM + ((lane >> 4) << 3);
#pragma unroll
    for (int s = 0; s < 4; ++s) {
        const int ncol = Nbase + s * 16 + am;
        float ps0 = 0.0f, ps1 = 0.0f;
#pragma unroll
        for (int r = 0; r < 8; ++r) {
            const int grow = mrowBase + r;
            if (grow < M_TOTAL) {
                const float val = acc[s][r];
                __builtin_nontemporal_store(val, &Lout[(size_t)grow * C_ + ncol]);
                if (grow / MB_ROWS == b0) ps0 += val; else ps1 += val;
            }
        }
        atomicAdd(&colSum[s * 16 + am], ps0);
        if (ps1 != 0.0f) atomicAdd(&colSum[64 + s * 16 + am], ps1);
    }
    __syncthreads();

    if (tid < 64) {
        atomicAdd(&sum_l[(size_t)b0 * C_ + Nbase + tid], colSum[tid]);
    } else if (tid < 128) {
        const int b1 = (Mbase + 127) / MB_ROWS;
        const float v = colSum[tid];
        if (b1 != b0 && b1 < B_ && v != 0.0f)
            atomicAdd(&sum_l[(size_t)b1 * C_ + Nbase + (tid - 64)], v);
    }
}

// ---------------------------------------------------------------------------
// mean = (sum_g + sum_l) / 14705; ghat = l2norm(gproj - mean)
__global__ void __launch_bounds__(256)
mean_ghat_kernel(const float* __restrict__ sum_g, const float* __restrict__ sum_l,
                 const float* __restrict__ gproj, float* __restrict__ mean,
                 float* __restrict__ ghat) {
    const int b = blockIdx.x;
    const int tid = threadIdx.x;
    __shared__ float mS[C_];
    __shared__ float red[256];
    for (int d = tid; d < C_; d += 256) {
        float m = (sum_g[b * C_ + d] + sum_l[b * C_ + d]) * (1.0f / (float)NTOK);
        mS[d] = m;
        mean[b * C_ + d] = m;
    }
    __syncthreads();
    for (int gi = 0; gi < NG_; ++gi) {
        const float* gr = gproj + (size_t)(b * NG_ + gi) * C_;
        float p = 0.0f;
        for (int d = tid; d < C_; d += 256) {
            const float x = gr[d] - mS[d];
            p = fmaf(x, x, p);
        }
        red[tid] = p;
        __syncthreads();
        for (int s = 128; s; s >>= 1) {
            if (tid < s) red[tid] += red[tid + s];
            __syncthreads();
        }
        const float rinv = __frsqrt_rn(red[0] + EPS_);
        for (int d = tid; d < C_; d += 256)
            ghat[(size_t)(b * NG_ + gi) * C_ + d] = (gr[d] - mS[d]) * rinv;
        __syncthreads();
    }
}

// ---------------------------------------------------------------------------
// Per (b,nl): center+normalize l rows on the fly, dot with 5 ghat vectors,
// softmax over nf. Memory-bound streaming re-read of l (301 MB total).
__global__ void __launch_bounds__(256)
attn_kernel(const float* __restrict__ lproj, const float* __restrict__ mean,
            const float* __restrict__ ghat, float* __restrict__ out) {
    const int blk = blockIdx.x;             // b*NL + li
    const int b  = blk / NL_;
    __shared__ float meanS[C_];
    __shared__ float gS[NG_ * C_];
    __shared__ float scoreS[NG_][200];
    const int tid  = threadIdx.x;
    const int wave = tid >> 5;
    const int lane = tid & 31;

    for (int i = tid; i < C_; i += 256) meanS[i] = mean[b * C_ + i];
    for (int i = tid; i < NG_ * C_; i += 256) gS[i] = ghat[(size_t)b * NG_ * C_ + i];
    __syncthreads();

    for (int f = wave; f < NF_; f += 8) {
        const float* lr = lproj + ((size_t)blk * NF_ + f) * C_;
        float nrm = 0.f, d0 = 0.f, d1 = 0.f, d2 = 0.f, d3 = 0.f, d4 = 0.f;
        for (int c = lane; c < C_; c += 32) {
            const float x = __builtin_nontemporal_load(&lr[c]) - meanS[c];
            nrm = fmaf(x, x, nrm);
            d0 = fmaf(x, gS[c],          d0);
            d1 = fmaf(x, gS[C_ + c],     d1);
            d2 = fmaf(x, gS[2 * C_ + c], d2);
            d3 = fmaf(x, gS[3 * C_ + c], d3);
            d4 = fmaf(x, gS[4 * C_ + c], d4);
        }
#pragma unroll
        for (int off = 16; off; off >>= 1) {
            nrm += __shfl_xor(nrm, off, 32);
            d0  += __shfl_xor(d0,  off, 32);
            d1  += __shfl_xor(d1,  off, 32);
            d2  += __shfl_xor(d2,  off, 32);
            d3  += __shfl_xor(d3,  off, 32);
            d4  += __shfl_xor(d4,  off, 32);
        }
        if (lane == 0) {
            const float rinv = __frsqrt_rn(nrm + EPS_) * ALPHA_INV;
            scoreS[0][f] = d0 * rinv;
            scoreS[1][f] = d1 * rinv;
            scoreS[2][f] = d2 * rinv;
            scoreS[3][f] = d3 * rinv;
            scoreS[4][f] = d4 * rinv;
        }
    }
    __syncthreads();

    if (wave < NG_) {
        const int g = wave;
        float m = -1e30f;
        for (int f = lane; f < NF_; f += 32) m = fmaxf(m, scoreS[g][f]);
#pragma unroll
        for (int off = 16; off; off >>= 1) m = fmaxf(m, __shfl_xor(m, off, 32));
        float s = 0.f;
        for (int f = lane; f < NF_; f += 32) s += __expf(scoreS[g][f] - m);
#pragma unroll
        for (int off = 16; off; off >>= 1) s += __shfl_xor(s, off, 32);
        const float inv = 1.0f / s;
        float* op = out + ((size_t)blk * NG_ + g) * NF_;
        for (int f = lane; f < NF_; f += 32)
            op[f] = __expf(scoreS[g][f] - m) * inv;
    }
}

// ---------------------------------------------------------------------------
extern "C" void kernel_launch(void* const* d_in, const int* in_sizes, int n_in,
                              void* d_out, int out_size, void* d_ws, size_t ws_size,
                              hipStream_t stream) {
    const float* global_f = (const float*)d_in[0];   // [8,5,640]
    const float* local_f  = (const float*)d_in[1];   // [8,75,196,640]
    const float* Wq       = (const float*)d_in[2];   // [640,640]
    const float* Wk       = (const float*)d_in[3];   // [640,640]
    float* out = (float*)d_out;                      // [8,75,5,196,1]

    // workspace layout (floats)
    float* ws     = (float*)d_ws;
    float* sum_g  = ws;                                   //  8*640
    float* sum_l  = ws + (size_t)B_ * C_;                 //  8*640
    float* mean   = ws + (size_t)2 * B_ * C_;             //  8*640
    float* gproj  = ws + (size_t)3 * B_ * C_;             //  8*5*640
    float* ghat   = gproj + (size_t)B_ * NG_ * C_;        //  8*5*640
    float* lproj  = ghat  + (size_t)B_ * NG_ * C_;        //  117600*640 (~301 MB)

    // 1) zero the atomic accumulators (ws is poisoned, not re-zeroed)
    zero_kernel<<<(2 * B_ * C_ + 255) / 256, 256, 0, stream>>>(sum_g, 2 * B_ * C_);

    // 2) tiny global projection + sums
    proj_global_kernel<<<B_ * NG_, 128, 0, stream>>>(global_f, Wq, gproj, sum_g);

    // 3) heavy WMMA GEMM + fused per-batch column sums
    dim3 g2((M_TOTAL + 127) / 128, C_ / 64);
    proj_local_wmma_kernel<<<g2, 256, 0, stream>>>(local_f, Wk, lproj, sum_l);

    // 4) mean + normalized global features
    mean_ghat_kernel<<<B_, 256, 0, stream>>>(sum_g, sum_l, gproj, mean, ghat);

    // 5) fused center/normalize/score/softmax
    attn_kernel<<<B_ * NL_, 256, 0, stream>>>(lproj, mean, ghat, out);
}
